// BiLSTMCRFSequenceTaggingModel_5299989643878
// MI455X (gfx1250) — compile-verified, matrix operand order checked by reading, not run on hardware
//
#include <hip/hip_runtime.h>
#include <hip/hip_bf16.h>

// ---------------------------------------------------------------------------
// BiLSTM-CRF forward on MI455X (gfx1250, wave32, WMMA bf16, TDM preload)
// ---------------------------------------------------------------------------
// Shapes: B=32, T=256, D=768, Hd=384, gates=1536, L=2 layers, C=14 classes.
// Pipeline: embed+LN -> per layer { inproj GEMM [WMMA, both dirs] ->
//           recurrent [16 WGs, LDS-resident W_hh via TENSOR_LOAD_TO_LDS,
//           WMMA, device-scope barrier] } -> FC(768->14) -> CRF -> loss.
// ---------------------------------------------------------------------------

typedef __bf16 bf16;
typedef __attribute__((ext_vector_type(16))) __bf16 v16bf;
typedef __attribute__((ext_vector_type(8)))  __bf16 v8bf;
typedef __attribute__((ext_vector_type(8)))  float  v8f;

typedef unsigned int u32x4 __attribute__((ext_vector_type(4)));
typedef int          i32x4 __attribute__((ext_vector_type(4)));
typedef int          i32x8 __attribute__((ext_vector_type(8)));

#define B_    32
#define T_    256
#define D_    768
#define HD_   384
#define NG_   1536
#define C_    14
#define NWGD_ 8            // workgroups per direction in recurrent kernel
#define JSL_  48           // hidden-column slice per WG (384/8)
#define NCOL_ 192          // gate columns per WG (4*48)

__device__ __forceinline__ float sigf(float x) { return 1.f / (1.f + __expf(-x)); }

// Load one 16x32 bf16 fragment (A or B operand) from a row-major matrix with
// leading dimension ldk, following the CDNA5 16-bit operand VGPR layout:
//   lanes 0-15 : row = row0+lane,    K = k0+{0..7}  and k0+{16..23}
//   lanes 16-31: row = row0+lane-16, K = k0+{8..15} and k0+{24..31}
template <typename P>
__device__ __forceinline__ v16bf load_frag(const P* base, int ldk, int row0, int k0, int lane) {
  const int laneLo = lane & 15;
  const int laneHi = (lane >> 4) & 1;
  const P* p = base + (size_t)(row0 + laneLo) * ldk + (k0 + laneHi * 8);
  v8bf lo = *(const v8bf*)(p);
  v8bf hi = *(const v8bf*)(p + 16);
  v16bf r;
#pragma unroll
  for (int i = 0; i < 8; ++i) { r[i] = lo[i]; r[i + 8] = hi[i]; }
  return r;
}

// Issue one TDM 2-D tile load (rows x 384 bf16 columns) from global to LDS.
// Descriptor layout per CDNA5 ISA ch.8 (D# groups 0/1; groups 2/3 zero = 2D).
__device__ __forceinline__ void tdm_load_2d(const bf16* gsrc, unsigned lds_byte_addr,
                                            unsigned rows) {
#if __has_builtin(__builtin_amdgcn_tensor_load_to_lds)
  const unsigned long long ga = (unsigned long long)(uintptr_t)gsrc;
  u32x4 g0;
  g0[0] = 1u;                                  // count=1, user mode
  g0[1] = lds_byte_addr;                       // D#.lds_addr
  g0[2] = (unsigned)(ga & 0xffffffffu);        // global_addr[31:0]
  g0[3] = (unsigned)((ga >> 32) & 0x01ffffffu) // global_addr[56:32]
          | (2u << 30);                        // type=2 ("image")
  i32x8 g1;
  g1[0] = (int)(1u << 16);                     // data_size=1 (2 bytes), mask=0
  g1[1] = (int)((HD_ & 0xffffu) << 16);        // tensor_dim0[15:0] @ bits 63:48
  g1[2] = (int)((HD_ >> 16) | ((rows & 0xffffu) << 16));  // dim0 hi | tensor_dim1 lo
  g1[3] = (int)((rows >> 16) | ((unsigned)HD_ << 16));    // dim1 hi | tile_dim0
  g1[4] = (int)rows;                           // tile_dim1 (tile_dim2=0)
  g1[5] = (int)HD_;                            // tensor_dim0_stride[31:0]
  g1[6] = 0;
  g1[7] = 0;
  i32x4 z4; z4[0] = z4[1] = z4[2] = z4[3] = 0;
#if __clang_major__ >= 23
  i32x8 z8; 
#pragma unroll
  for (int i = 0; i < 8; ++i) z8[i] = 0;
  __builtin_amdgcn_tensor_load_to_lds(g0, g1, z4, z4, z8, 0);
#else
  __builtin_amdgcn_tensor_load_to_lds(g0, g1, z4, z4, 0);
#endif
#else
  (void)gsrc; (void)lds_byte_addr; (void)rows;
#endif
}

// ---------------------------------------------------------------------------
// 0) f32 -> bf16 weight conversion
// ---------------------------------------------------------------------------
__global__ __launch_bounds__(256) void cvt_bf16_kernel(const float* __restrict__ in,
                                                       bf16* __restrict__ out, int n) {
  int i = blockIdx.x * 256 + threadIdx.x;
  if (i < n) out[i] = (bf16)in[i];
}

// Zero the recurrent h double-buffers and the grid-barrier counter
// (re-run every layer so graph replays are deterministic).
__global__ __launch_bounds__(256) void init_state_kernel(bf16* __restrict__ Hbuf,
                                                         unsigned* __restrict__ bar) {
  int i = blockIdx.x * 256 + threadIdx.x;
  if (i < 2 * 2 * B_ * HD_) Hbuf[i] = (bf16)0.f;
  if (i == 0) bar[0] = 0u;
}

// ---------------------------------------------------------------------------
// 1) Embedding gather + LayerNorm -> X[t*B + b][768] in bf16 (time-major)
// ---------------------------------------------------------------------------
__global__ __launch_bounds__(256) void embed_ln_kernel(
    const int* __restrict__ ids, const float* __restrict__ we,
    const float* __restrict__ pe, const float* __restrict__ te,
    const float* __restrict__ g, const float* __restrict__ bb,
    bf16* __restrict__ X) {
  __shared__ float buf[D_];
  __shared__ float red[256];
  const int row = blockIdx.x;          // row = t*B + b
  const int t = row / B_, b = row % B_;
  const int tid = threadIdx.x;
  const int id = ids[b * T_ + t];

  float lsum = 0.f;
  for (int d = tid; d < D_; d += 256) {
    float v = we[(size_t)id * D_ + d] + pe[(size_t)t * D_ + d] + te[d];
    buf[d] = v;
    lsum += v;
  }
  red[tid] = lsum;
  __syncthreads();
  for (int s = 128; s > 0; s >>= 1) {
    if (tid < s) red[tid] += red[tid + s];
    __syncthreads();
  }
  const float mu = red[0] * (1.f / (float)D_);
  __syncthreads();

  float lsq = 0.f;
  for (int d = tid; d < D_; d += 256) {
    float v = buf[d] - mu;
    lsq += v * v;
  }
  red[tid] = lsq;
  __syncthreads();
  for (int s = 128; s > 0; s >>= 1) {
    if (tid < s) red[tid] += red[tid + s];
    __syncthreads();
  }
  const float rstd = rsqrtf(red[0] * (1.f / (float)D_) + 1e-12f);

  bf16* xr = X + (size_t)row * D_;
  for (int d = tid; d < D_; d += 256)
    xr[d] = (bf16)((buf[d] - mu) * rstd * g[d] + bb[d]);
}

// ---------------------------------------------------------------------------
// 2) Input projection GEMM (both directions via blockIdx.y):
//    G[8192,1536] = X[8192,768] * Wih^T + (b_ih+b_hh)
//    bf16 WMMA, f32 accumulate. Each wave owns a 64x64 output block.
// ---------------------------------------------------------------------------
__global__ __launch_bounds__(256) void inproj_gemm_kernel(
    const bf16* __restrict__ X,         // [T*B, 768]
    const bf16* __restrict__ Wbase,     // [2, 1536, 768] row-major (this layer)
    const float* __restrict__ bihBase,  // [2, 1536]
    const float* __restrict__ bhhBase,  // [2, 1536]
    float* __restrict__ Gf, float* __restrict__ Gr) {
  const int d = blockIdx.y;
  const bf16* W = Wbase + (size_t)d * NG_ * D_;
  const float* bih = bihBase + d * NG_;
  const float* bhh = bhhBase + d * NG_;
  float* G = d ? Gr : Gf;

  const int wave = threadIdx.x >> 5;
  const int lane = threadIdx.x & 31;
  const int gw = blockIdx.x * 8 + wave;
  const int NT = NG_ / 64;  // 24 tiles of 64 cols
  const int ty = gw / NT, tx = gw % NT;
  const int m0 = ty * 64, n0 = tx * 64;

  v8f acc[4][4];
#pragma unroll
  for (int i = 0; i < 4; ++i)
#pragma unroll
    for (int j = 0; j < 4; ++j)
#pragma unroll
      for (int r = 0; r < 8; ++r) acc[i][j][r] = 0.f;

  for (int k0 = 0; k0 < D_; k0 += 32) {
    v16bf a[4];
#pragma unroll
    for (int i = 0; i < 4; ++i) a[i] = load_frag(X, D_, m0 + 16 * i, k0, lane);
#pragma unroll
    for (int j = 0; j < 4; ++j) {
      v16bf b = load_frag(W, D_, n0 + 16 * j, k0, lane);
#pragma unroll
      for (int i = 0; i < 4; ++i)
        acc[i][j] = __builtin_amdgcn_wmma_f32_16x16x32_bf16(
            false, a[i], false, b, (short)0, acc[i][j], false, false);
    }
  }

  const int laneLo = lane & 15;
  const int laneHi = (lane >> 4) & 1;
#pragma unroll
  for (int j = 0; j < 4; ++j) {
    const int col = n0 + 16 * j + laneLo;
    const float bias = bih[col] + bhh[col];
#pragma unroll
    for (int i = 0; i < 4; ++i) {
      const int rb = m0 + 16 * i + laneHi * 8;
#pragma unroll
      for (int r = 0; r < 8; ++r)
        G[(size_t)(rb + r) * NG_ + col] = acc[i][j][r] + bias;
    }
  }
}

// ---------------------------------------------------------------------------
// 3) Recurrent LSTM, 16 workgroups (8 per direction), 256 threads each.
//    WG `wg` of direction `dir` owns gate columns {g*384 + wg*48 + jl} for
//    g=0..3, jl=0..47. Its 192 W_hh rows (147 KB bf16) are DMA'd into LDS
//    once per layer by the Tensor Data Mover (4 x 48-row 2D tiles); per step
//    it reads only h (24 KB) + its G slice from L2. h state is
//    double-buffered in global memory; steps are separated by a device-scope
//    atomic barrier (s_sleep backoff). c stays in registers.
//    Dynamic LDS: 147456 (W) + 24576 (h) + 24576 (g) = 196608 B (< 320 KB).
// ---------------------------------------------------------------------------
__global__ __launch_bounds__(256) void lstm_recur_kernel(
    const float* __restrict__ Gf, const float* __restrict__ Gr,  // [T*B, 1536]
    const bf16* __restrict__ Wf, const bf16* __restrict__ Wr,    // [1536, 384]
    bf16* __restrict__ Hbuf,        // [2 dirs][2 bufs][32*384]
    unsigned* __restrict__ bar,     // grid barrier counter (pre-zeroed)
    bf16* __restrict__ Xout) {      // [T*B, 768]
  extern __shared__ char smem[];
  bf16*  w_lds = (bf16*)smem;                                   // [192][384]
  bf16*  h_lds = (bf16*)(smem + (size_t)NCOL_ * HD_ * 2);       // [32][384]
  float* g_lds = (float*)(smem + (size_t)NCOL_ * HD_ * 2 + (size_t)B_ * HD_ * 2); // [32][192]

  const int dir = blockIdx.x >> 3;
  const int wg  = blockIdx.x & 7;
  const float* G = dir ? Gr : Gf;
  const bf16* W = dir ? Wr : Wf;
  bf16* Hd = Hbuf + (size_t)dir * 2 * (B_ * HD_);

  const int tid = threadIdx.x;
  const int wave = tid >> 5;
  const int lane = tid & 31;
  const int laneLo = lane & 15;
  const int laneHi = (lane >> 4) & 1;

  // Preload this WG's W_hh slice into LDS (192 rows x 384 cols bf16).
#if __has_builtin(__builtin_amdgcn_tensor_load_to_lds)
  if (wave == 0) {
    const unsigned ldsBase = (unsigned)(uintptr_t)(void*)w_lds;  // addr[31:0] = LDS offset
#pragma unroll
    for (int g = 0; g < 4; ++g) {
      const bf16* gsrc = W + ((size_t)g * HD_ + (size_t)wg * JSL_) * HD_;
      tdm_load_2d(gsrc, ldsBase + (unsigned)g * JSL_ * HD_ * 2, JSL_);
    }
    __builtin_amdgcn_s_wait_tensorcnt(0);
  }
#else
  {
    const uint4* __restrict__ src = (const uint4*)W;  // 48 uint4 per row
    uint4* dst = (uint4*)w_lds;
    for (int idx = tid; idx < NCOL_ * 48; idx += 256) {
      const int r = idx / 48, c = idx % 48;
      const int g = r / JSL_, jl = r % JSL_;
      const int grow = g * HD_ + wg * JSL_ + jl;
      dst[r * 48 + c] = src[(size_t)grow * 48 + c];
    }
  }
#endif
  // h(0) = 0
  {
    uint4 z; z.x = z.y = z.z = z.w = 0u;
    uint4* dst = (uint4*)h_lds;
    for (int i = tid; i < (B_ * HD_) / 8; i += 256) dst[i] = z;
  }
  float c_reg[6];
#pragma unroll
  for (int q = 0; q < 6; ++q) c_reg[q] = 0.f;
  __syncthreads();

  for (int step = 0; step < T_; ++step) {
    const int t = dir ? (T_ - 1 - step) : step;

    // 24 output tiles (2 M x 12 N), 3 per wave; operands fully in LDS.
#pragma unroll
    for (int q = 0; q < 3; ++q) {
      const int tl = wave + q * 8;
      const int m0 = (tl / 12) * 16;
      const int n0 = (tl % 12) * 16;
      v8f acc = {0.f, 0.f, 0.f, 0.f, 0.f, 0.f, 0.f, 0.f};
#pragma unroll
      for (int k0 = 0; k0 < HD_; k0 += 32) {
        v16bf a = load_frag(h_lds, HD_, m0, k0, lane);
        v16bf b = load_frag(w_lds, HD_, n0, k0, lane);
        acc = __builtin_amdgcn_wmma_f32_16x16x32_bf16(
            false, a, false, b, (short)0, acc, false, false);
      }
      const int col = n0 + laneLo;
      const int rb = m0 + laneHi * 8;
#pragma unroll
      for (int r = 0; r < 8; ++r) g_lds[(rb + r) * NCOL_ + col] = acc[r];
    }
    __syncthreads();

    // Fused gates: 32*48 = 1536 (b, jl) pairs, 6 per thread.
    const float* Grow = G + (size_t)t * B_ * NG_;
    bf16* Hnext = Hd + ((step + 1) & 1) * (B_ * HD_);
#pragma unroll
    for (int q = 0; q < 6; ++q) {
      const int p = tid + q * 256;
      const int b = p / JSL_, jl = p % JSL_;
      const int jg = wg * JSL_ + jl;
      const float gi = g_lds[b * NCOL_ + jl]            + Grow[(size_t)b * NG_ + jg];
      const float gf = g_lds[b * NCOL_ + JSL_ + jl]     + Grow[(size_t)b * NG_ + HD_ + jg];
      const float gg = g_lds[b * NCOL_ + 2 * JSL_ + jl] + Grow[(size_t)b * NG_ + 2 * HD_ + jg];
      const float go = g_lds[b * NCOL_ + 3 * JSL_ + jl] + Grow[(size_t)b * NG_ + 3 * HD_ + jg];
      const float c = sigf(gf) * c_reg[q] + sigf(gi) * tanhf(gg);
      c_reg[q] = c;
      const float h = sigf(go) * tanhf(c);
      Hnext[b * HD_ + jg] = (bf16)h;
      Xout[((size_t)t * B_ + b) * D_ + dir * HD_ + jg] = (bf16)h;
    }

    // Device-scope barrier across all 16 WGs (both directions in lockstep).
    __syncthreads();
    if (tid == 0) {
      __threadfence();
      atomicAdd(bar, 1u);
      const unsigned target = (unsigned)(step + 1) * (2u * NWGD_);
      while (__hip_atomic_load(bar, __ATOMIC_ACQUIRE, __HIP_MEMORY_SCOPE_AGENT) < target)
        __builtin_amdgcn_s_sleep(1);
      __threadfence();
    }
    __syncthreads();

    // Refresh h_lds <- h(step+1) from the just-published buffer.
    if (step + 1 < T_) {
      const uint4* src = (const uint4*)(Hd + ((step + 1) & 1) * (B_ * HD_));
      uint4* dst = (uint4*)h_lds;
      for (int i = tid; i < (B_ * HD_) / 8; i += 256) dst[i] = src[i];
      __syncthreads();
    }
  }
}

// ---------------------------------------------------------------------------
// 4) FC emissions: em[b][t][14] = X[t*B+b] . fc_w[c] + fc_b[c]
// ---------------------------------------------------------------------------
__global__ __launch_bounds__(256) void fc_kernel(
    const bf16* __restrict__ X, const float* __restrict__ fw,
    const float* __restrict__ fb, float* __restrict__ em) {
  const int wave = threadIdx.x >> 5;
  const int lane = threadIdx.x & 31;
  const int row = blockIdx.x * 8 + wave;  // row = t*B + b
  const int t = row / B_, b = row % B_;
  const bf16* x = X + (size_t)row * D_;

  float p[C_];
#pragma unroll
  for (int c = 0; c < C_; ++c) p[c] = 0.f;
  for (int k = lane; k < D_; k += 32) {
    const float xv = (float)x[k];
#pragma unroll
    for (int c = 0; c < C_; ++c) p[c] += xv * fw[c * D_ + k];
  }
#pragma unroll
  for (int c = 0; c < C_; ++c) {
    float v = p[c];
    for (int o = 16; o > 0; o >>= 1) v += __shfl_xor(v, o, 32);
    if (lane == 0) em[((size_t)b * T_ + t) * C_ + c] = v + fb[c];
  }
}

// ---------------------------------------------------------------------------
// 5) CRF: one block, wave w <-> batch b, lane j <-> state j (j < 14).
// ---------------------------------------------------------------------------
__global__ __launch_bounds__(1024) void crf_kernel(
    const float* __restrict__ em, const int* __restrict__ labels,
    const float* __restrict__ cs, const float* __restrict__ ce,
    const float* __restrict__ ct, float* __restrict__ out) {
  __shared__ float trans_s[C_ * C_];
  __shared__ float partial[32];
  const int tid = threadIdx.x;
  const int wave = tid >> 5;
  const int lane = tid & 31;
  if (tid < C_ * C_) trans_s[tid] = ct[tid];
  __syncthreads();

  const int b = wave;
  const float* E = em + (size_t)b * T_ * C_;
  const int* Y = labels + (size_t)b * T_;

  float sc = 0.f;
  for (int t = lane; t < T_; t += 32) {
    const int yt = Y[t];
    sc += E[t * C_ + yt];
    if (t > 0) sc += trans_s[Y[t - 1] * C_ + yt];
  }
  for (int o = 16; o > 0; o >>= 1) sc += __shfl_xor(sc, o, 32);
  sc += cs[Y[0]] + ce[Y[T_ - 1]];

  float alpha = (lane < C_) ? (cs[lane] + E[lane]) : -1e30f;
  for (int t = 1; t < T_; ++t) {
    float vals[C_];
    float m = -1e30f;
#pragma unroll
    for (int i = 0; i < C_; ++i) {
      const float ai = __shfl(alpha, i, 32);
      const float v = ai + ((lane < C_) ? trans_s[i * C_ + lane] : 0.f);
      vals[i] = v;
      m = fmaxf(m, v);
    }
    float s = 0.f;
#pragma unroll
    for (int i = 0; i < C_; ++i) s += __expf(vals[i] - m);
    const float anew = m + __logf(s) + ((lane < C_) ? E[t * C_ + lane] : 0.f);
    alpha = (lane < C_) ? anew : -1e30f;
  }

  float v = (lane < C_) ? (alpha + ce[lane]) : -1e30f;
  float m = v;
  for (int o = 16; o > 0; o >>= 1) m = fmaxf(m, __shfl_xor(m, o, 32));
  float s = (lane < C_) ? __expf(v - m) : 0.f;
  for (int o = 16; o > 0; o >>= 1) s += __shfl_xor(s, o, 32);
  const float logZ = m + __logf(s);

  if (lane == 0) partial[wave] = logZ - sc;
  __syncthreads();
  if (tid == 0) {
    float tot = 0.f;
    for (int i = 0; i < 32; ++i) tot += partial[i];
    out[0] = tot;
  }
}

// ---------------------------------------------------------------------------
// Host-side orchestration
// ---------------------------------------------------------------------------
extern "C" void kernel_launch(void* const* d_in, const int* in_sizes, int n_in,
                              void* d_out, int out_size, void* d_ws, size_t ws_size,
                              hipStream_t stream) {
  (void)in_sizes; (void)n_in; (void)out_size; (void)ws_size;
  const int*   input_ids = (const int*)d_in[0];
  const int*   labels    = (const int*)d_in[1];
  const float* word_emb  = (const float*)d_in[2];
  const float* pos_emb   = (const float*)d_in[3];
  const float* type_emb  = (const float*)d_in[4];
  const float* ln_g      = (const float*)d_in[5];
  const float* ln_b      = (const float*)d_in[6];
  const float* w_ih      = (const float*)d_in[7];   // [2,2,1536,768]
  const float* w_hh      = (const float*)d_in[8];   // [2,2,1536,384]
  const float* b_ih      = (const float*)d_in[9];   // [2,2,1536]
  const float* b_hh      = (const float*)d_in[10];
  const float* fc_w      = (const float*)d_in[11];  // [14,768]
  const float* fc_b      = (const float*)d_in[12];
  const float* crf_start = (const float*)d_in[13];
  const float* crf_end   = (const float*)d_in[14];
  const float* crf_trans = (const float*)d_in[15];
  float* out = (float*)d_out;

  // Workspace layout (256-byte aligned slabs)
  char* ws = (char*)d_ws;
  size_t off = 0;
  auto take = [&](size_t bytes) {
    char* p = ws + off;
    off = (off + bytes + 255) & ~(size_t)255;
    return p;
  };
  const size_t nWih = (size_t)2 * 2 * NG_ * D_;    // 4,718,592
  const size_t nWhh = (size_t)2 * 2 * NG_ * HD_;   // 2,359,296
  bf16*     Wih_bf = (bf16*)take(nWih * sizeof(bf16));
  bf16*     Whh_bf = (bf16*)take(nWhh * sizeof(bf16));
  bf16*     Xa     = (bf16*)take((size_t)T_ * B_ * D_ * sizeof(bf16));
  bf16*     Xb     = (bf16*)take((size_t)T_ * B_ * D_ * sizeof(bf16));
  float*    Gf     = (float*)take((size_t)T_ * B_ * NG_ * sizeof(float));
  float*    Gr     = (float*)take((size_t)T_ * B_ * NG_ * sizeof(float));
  float*    em     = (float*)take((size_t)B_ * T_ * C_ * sizeof(float));
  bf16*     Hbuf   = (bf16*)take((size_t)2 * 2 * B_ * HD_ * sizeof(bf16));
  unsigned* bar    = (unsigned*)take(256);

  // 0) weights -> bf16
  cvt_bf16_kernel<<<(unsigned)((nWih + 255) / 256), 256, 0, stream>>>(w_ih, Wih_bf, (int)nWih);
  cvt_bf16_kernel<<<(unsigned)((nWhh + 255) / 256), 256, 0, stream>>>(w_hh, Whh_bf, (int)nWhh);

  // 1) embeddings + LayerNorm -> Xa (time-major)
  embed_ln_kernel<<<T_ * B_, 256, 0, stream>>>(input_ids, word_emb, pos_emb,
                                               type_emb, ln_g, ln_b, Xa);

  // 2) two BiLSTM layers
  bf16* xin = Xa;
  bf16* xout = Xb;
  const size_t szWih = (size_t)NG_ * D_;
  const size_t szWhh = (size_t)NG_ * HD_;
  const dim3 inprojGrid((unsigned)(((T_ * B_) / 64) * (NG_ / 64) / 8), 2);  // (384, 2)
  const size_t recurLds = (size_t)NCOL_ * HD_ * sizeof(bf16)   // W slice
                        + (size_t)B_ * HD_ * sizeof(bf16)      // h
                        + (size_t)B_ * NCOL_ * sizeof(float);  // gates
  for (int l = 0; l < 2; ++l) {
    inproj_gemm_kernel<<<inprojGrid, 256, 0, stream>>>(
        xin, Wih_bf + (size_t)(l * 2) * szWih,
        b_ih + (size_t)(l * 2) * NG_, b_hh + (size_t)(l * 2) * NG_, Gf, Gr);
    init_state_kernel<<<(2 * 2 * B_ * HD_ + 255) / 256, 256, 0, stream>>>(Hbuf, bar);
    lstm_recur_kernel<<<2 * NWGD_, 256, recurLds, stream>>>(
        Gf, Gr, Whh_bf + (size_t)(l * 2 + 0) * szWhh,
        Whh_bf + (size_t)(l * 2 + 1) * szWhh, Hbuf, bar, xout);
    bf16* tmp = xin; xin = xout; xout = tmp;
  }

  // 3) emissions
  fc_kernel<<<(T_ * B_) / 8, 256, 0, stream>>>(xin, fc_w, fc_b, em);

  // 4) CRF loss (single deterministic write to out[0])
  crf_kernel<<<1, 1024, 0, stream>>>(em, labels, crf_start, crf_end, crf_trans, out);
}
